// AffineTransform_24146306138541
// MI455X (gfx1250) — matrix operand order repaired and checked by reading, not compiled, not run on hardware
//
#include <hip/hip_runtime.h>

typedef __attribute__((ext_vector_type(16))) _Float16 v16h;
typedef __attribute__((ext_vector_type(8)))  _Float16 v8h;
typedef __attribute__((ext_vector_type(8)))  float    v8f;

#define NTOK  256
#define DHEAD 32
#define NHEAD 8
#define NPOS  961   // (2*16-1)^2

// ---------------------------------------------------------------------------
// Kernel 1: relative-position MLP  (961 rows: 2 -> 16 -> 16 -> 16 -> 8)
// ---------------------------------------------------------------------------
__device__ __forceinline__ void ln_relu16(const float* x, const float* g,
                                          const float* b, float* y) {
  float m = 0.f;
  #pragma unroll
  for (int i = 0; i < 16; ++i) m += x[i];
  m *= (1.0f / 16.0f);
  float v = 0.f;
  #pragma unroll
  for (int i = 0; i < 16; ++i) { float d = x[i] - m; v += d * d; }
  v *= (1.0f / 16.0f);
  float rs = rsqrtf(v + 1e-5f);
  #pragma unroll
  for (int i = 0; i < 16; ++i) {
    float t = (x[i] - m) * rs * g[i] + b[i];
    y[i] = t > 0.f ? t : 0.f;
  }
}

__global__ void pos_mlp_kernel(const float* __restrict__ wp, const float* __restrict__ bp,
                               const float* __restrict__ g1, const float* __restrict__ e1,
                               const float* __restrict__ w1, const float* __restrict__ b1,
                               const float* __restrict__ g2, const float* __restrict__ e2,
                               const float* __restrict__ w2, const float* __restrict__ b2,
                               const float* __restrict__ g3, const float* __restrict__ e3,
                               const float* __restrict__ w3, const float* __restrict__ b3,
                               float* __restrict__ pos) {
  int i = blockIdx.x * blockDim.x + threadIdx.x;
  if (i >= NPOS) return;
  float fh = (float)(i / 31 - 15);
  float fw = (float)(i % 31 - 15);
  float x[16], y[16];
  #pragma unroll
  for (int j = 0; j < 16; ++j) x[j] = fh * wp[j] + fw * wp[16 + j] + bp[j];
  ln_relu16(x, g1, e1, y);
  #pragma unroll
  for (int k = 0; k < 16; ++k) {
    float t = b1[k];
    #pragma unroll
    for (int j = 0; j < 16; ++j) t += y[j] * w1[j * 16 + k];
    x[k] = t;
  }
  ln_relu16(x, g2, e2, y);
  #pragma unroll
  for (int k = 0; k < 16; ++k) {
    float t = b2[k];
    #pragma unroll
    for (int j = 0; j < 16; ++j) t += y[j] * w2[j * 16 + k];
    x[k] = t;
  }
  ln_relu16(x, g3, e3, y);
  #pragma unroll
  for (int k = 0; k < NHEAD; ++k) {
    float t = b3[k];
    #pragma unroll
    for (int j = 0; j < 16; ++j) t += y[j] * w3[j * NHEAD + k];
    pos[i * NHEAD + k] = t;
  }
}

// ---------------------------------------------------------------------------
// Kernel 2: materialize bias directly in WMMA C-tile layout:
//   biasT[h][tileM][tileN][lane][r]  (lane = 32 wave lanes, r = 8 C VGPRs)
// so the attention kernel fetches each tile's bias as one contiguous v8f.
// C layout: query row = tileM*16 + r + 8*(lane>>4), key col = tileN*16 + (lane&15)
// ---------------------------------------------------------------------------
__global__ void bias_gather_kernel(const float* __restrict__ pos,
                                   float* __restrict__ biasT) {
  int t    = blockIdx.x * blockDim.x + threadIdx.x;  // h:3 | tm:4 | tn:4 | lane:5 | r:3
  int r    = t & 7;
  int lane = (t >> 3) & 31;
  int tn   = (t >> 8) & 15;
  int tm   = (t >> 12) & 15;
  int hh   = t >> 16;
  int qrow = tm * 16 + r + 8 * (lane >> 4);
  int kcol = tn * 16 + (lane & 15);
  int d0   = (qrow >> 4) - (kcol >> 4) + 15;   // row delta + (h-1)
  int d1   = (qrow & 15) - (kcol & 15) + 15;   // col delta + (w-1)
  biasT[t] = pos[(d0 * 31 + d1) * NHEAD + hh];
}

// ---------------------------------------------------------------------------
// Kernel 3: attention. One block = one (b,h) head. 512 threads = 16 waves.
// Wave w handles query rows [16w, 16w+16). All WMMAs are f16 in / f32 acc,
// bias folded in through the WMMA C operand.
// ---------------------------------------------------------------------------
__launch_bounds__(512, 1)
__global__ void attn_kernel(const float* __restrict__ q, const float* __restrict__ k,
                            const float* __restrict__ v, const float* __restrict__ biasT,
                            float* __restrict__ out) {
  // 48 KB static LDS
  __shared__ __attribute__((aligned(32))) _Float16 sK[NTOK * DHEAD];   // [tok][d]   16 KB
  __shared__ __attribute__((aligned(32))) _Float16 sVT[DHEAD * NTOK];  // [d][tok]   16 KB
  __shared__ __attribute__((aligned(32))) _Float16 sP[16 * 16 * DHEAD];// per-wave 1KB slices

  const int bh   = blockIdx.x;        // b*NHEAD + h
  const int h    = bh & (NHEAD - 1);
  const int tid  = threadIdx.x;
  const int wave = tid >> 5;
  const int lane = tid & 31;
  const int half = lane >> 4;         // 0: lanes 0-15, 1: lanes 16-31
  const int ln   = lane & 15;

  const float* __restrict__ qh = q + (size_t)bh * NTOK * DHEAD;
  const float* __restrict__ kh = k + (size_t)bh * NTOK * DHEAD;
  const float* __restrict__ vh = v + (size_t)bh * NTOK * DHEAD;
  // bias for this head, tiled: [tileM][tileN][lane][8]
  const float* __restrict__ bi = biasT + (size_t)h * (16 * 16 * 32 * 8);

  // ---- stage K (row major) and V (transposed) into LDS as f16 ----
  #pragma unroll
  for (int i = 0; i < 16; ++i) {
    int e   = i * 512 + tid;          // 8192 elements total
    int tok = e >> 5, d = e & 31;
    sK[e] = (_Float16)kh[e];
    sVT[d * NTOK + tok] = (_Float16)vh[e];
  }
  __syncthreads();

  // ---- Q A-tile: lane row = rowbase+ln; K-run {d0..d0+15} ----
  const int   rowbase = wave * 16;
  const int   d0      = half * 16;
  const float scale   = 0.17677669529663687f;  // 1/sqrt(32)
  const float* qr = qh + (rowbase + ln) * DHEAD;
  __attribute__((aligned(16))) float qt[16];
  *(float4*)&qt[0]  = *(const float4*)(qr + d0);
  *(float4*)&qt[4]  = *(const float4*)(qr + d0 + 4);
  *(float4*)&qt[8]  = *(const float4*)(qr + d0 + 8);
  *(float4*)&qt[12] = *(const float4*)(qr + d0 + 12);
  v16h qa;
  #pragma unroll
  for (int j = 0; j < 16; ++j) qa[j] = (_Float16)(qt[j] * scale);

  // ---- GEMM1: S = (Q*scale) K^T + bias, 16 tiles of 16x16 ----
  // bias enters through the WMMA C operand (D = A*B + C), loaded as one v8f.
  const float* __restrict__ biW = bi + ((size_t)wave * 16 * 32 + lane) * 8;
  v8f sacc[16];
  #pragma unroll
  for (int jn = 0; jn < 16; ++jn) {
    // B = K^T tile: lane holds column (key token jn*16+ln), contiguous d-run
    const v16h bk = *(const v16h*)&sK[(jn * 16 + ln) * DHEAD + 16 * half];
    const v8f  cb = *(const v8f*)(biW + (size_t)jn * 32 * 8);  // 32B coalesced
    sacc[jn] = __builtin_amdgcn_wmma_f32_16x16x32_f16(false, qa, false, bk,
                                                      (short)0, cb, false, false);
  }

  // ---- softmax over the 256 columns (in-lane over tiles + 16-lane shuffles) ----
  float inv[8];
  #pragma unroll
  for (int r = 0; r < 8; ++r) {
    float t = sacc[0][r];
    #pragma unroll
    for (int jn = 1; jn < 16; ++jn) t = fmaxf(t, sacc[jn][r]);
    t = fmaxf(t, __shfl_xor(t, 1));
    t = fmaxf(t, __shfl_xor(t, 2));
    t = fmaxf(t, __shfl_xor(t, 4));
    t = fmaxf(t, __shfl_xor(t, 8));
    float s = 0.f;
    #pragma unroll
    for (int jn = 0; jn < 16; ++jn) {
      float p = __expf(sacc[jn][r] - t);
      sacc[jn][r] = p;
      s += p;
    }
    s += __shfl_xor(s, 1);
    s += __shfl_xor(s, 2);
    s += __shfl_xor(s, 4);
    s += __shfl_xor(s, 8);
    inv[r] = 1.0f / s;
  }

  // ---- GEMM2: O = P V via C-layout -> A-layout transpose through LDS ----
  _Float16* __restrict__ pw = &sP[wave * (16 * DHEAD)];  // 16 rows x 32 k
  v8f o0 = {}, o1 = {};
  #pragma unroll
  for (int kb = 0; kb < 8; ++kb) {
    // write normalized P slice (k = kb*32 .. kb*32+31) from score tiles 2kb, 2kb+1
    #pragma unroll
    for (int t2 = 0; t2 < 2; ++t2) {
      const int jn = kb * 2 + t2;
      #pragma unroll
      for (int r = 0; r < 8; ++r)
        pw[(r + 8 * half) * DHEAD + t2 * 16 + ln] =
            (_Float16)(sacc[jn][r] * inv[r]);
    }
    asm volatile("s_wait_dscnt 0" ::: "memory");  // wave-private RAW fence on LDS
    // A-tile: lane row = ln; K-run {d0..d0+15}
    const v8h plo = *(const v8h*)&pw[ln * DHEAD + d0];
    const v8h phi = *(const v8h*)&pw[ln * DHEAD + 8 + d0];
    v16h pa;
    #pragma unroll
    for (int i = 0; i < 8; ++i) { pa[i] = plo[i]; pa[8 + i] = phi[i]; }
    // B-tiles from V^T: lane holds output dim column, contiguous token-run
    const v16h bv0 = *(const v16h*)&sVT[ln * NTOK + kb * 32 + 16 * half];
    const v16h bv1 = *(const v16h*)&sVT[(16 + ln) * NTOK + kb * 32 + 16 * half];
    o0 = __builtin_amdgcn_wmma_f32_16x16x32_f16(false, pa, false, bv0,
                                                (short)0, o0, false, false);
    o1 = __builtin_amdgcn_wmma_f32_16x16x32_f16(false, pa, false, bv1,
                                                (short)0, o1, false, false);
  }

  // ---- store O (C-layout -> coalesced f32 stores) ----
  float* __restrict__ oh = out + (size_t)bh * NTOK * DHEAD;
  #pragma unroll
  for (int r = 0; r < 8; ++r) {
    const int m = rowbase + r + 8 * half;
    oh[m * DHEAD + ln]      = o0[r];
    oh[m * DHEAD + 16 + ln] = o1[r];
  }
}

// ---------------------------------------------------------------------------
extern "C" void kernel_launch(void* const* d_in, const int* in_sizes, int n_in,
                              void* d_out, int out_size, void* d_ws, size_t ws_size,
                              hipStream_t stream) {
  const float* q  = (const float*)d_in[0];
  const float* k  = (const float*)d_in[1];
  const float* v  = (const float*)d_in[2];
  const float* wp = (const float*)d_in[3];
  const float* bp = (const float*)d_in[4];
  const float* g1 = (const float*)d_in[5];
  const float* e1 = (const float*)d_in[6];
  const float* w1 = (const float*)d_in[7];
  const float* b1 = (const float*)d_in[8];
  const float* g2 = (const float*)d_in[9];
  const float* e2 = (const float*)d_in[10];
  const float* w2 = (const float*)d_in[11];
  const float* b2 = (const float*)d_in[12];
  const float* g3 = (const float*)d_in[13];
  const float* e3 = (const float*)d_in[14];
  const float* w3 = (const float*)d_in[15];
  const float* b3 = (const float*)d_in[16];
  // d_in[17]=h, d_in[18]=w : fixed at 16 for these shapes

  float* pos   = (float*)d_ws;                         // 961*8 floats
  float* biasT = (float*)((char*)d_ws + 32768);        // 8*16*16*32*8 floats (2 MB)

  pos_mlp_kernel<<<8, 128, 0, stream>>>(wp, bp, g1, e1, w1, b1, g2, e2, w2, b2,
                                        g3, e3, w3, b3, pos);
  bias_gather_kernel<<<(NHEAD * NTOK * NTOK) / 256, 256, 0, stream>>>(pos, biasT);
  attn_kernel<<<128 * NHEAD, 512, 0, stream>>>(q, k, v, biasT, (float*)d_out);
}
